// LocalAttention_34729105555535
// MI455X (gfx1250) — matrix operand order repaired and checked by reading, not compiled, not run on hardware
//
#include <hip/hip_runtime.h>
#include <hip/hip_bf16.h>

typedef __attribute__((ext_vector_type(16))) __bf16 v16bf;
typedef __attribute__((ext_vector_type(8)))  float  v8f;

struct U8 { unsigned u[8]; };

__device__ __forceinline__ unsigned short f2bf(float f) {
  unsigned u = __builtin_bit_cast(unsigned, f);
  u += 0x7FFFu + ((u >> 16) & 1u);          // round-to-nearest-even
  return (unsigned short)(u >> 16);
}

#define LDSTRIDE 72   // bf16 elements per LDS row (64 + pad, keeps 8B alignment)

__global__ __launch_bounds__(512)
void local_attn_kernel(const float* __restrict__ q,
                       const float* __restrict__ k,
                       float* __restrict__ out) {
  __shared__ unsigned short Qs[128 * LDSTRIDE];   // 18 KB
  __shared__ unsigned short Ks[384 * LDSTRIDE];   // 54 KB
  __shared__ float redmax[2][128];
  __shared__ float redsum[2][128];

  const int o  = blockIdx.x;   // 0..61 output window index
  const int bh = blockIdx.y;   // 0..31
  int wi = o;                  // map to real window id, skipping excluded {2,34}
  if (wi >= 2)  ++wi;
  if (wi >= 34) ++wi;

  const int tid = threadIdx.x; // 0..511

  // ---------------- Stage Q (scaled by d^-1/2 = 0.125) as bf16 ----------------
  {
    const float4* qb =
        (const float4*)(q + (((size_t)bh * 8192) + (size_t)wi * 128) * 64);
    #pragma unroll
    for (int i = 0; i < 4; ++i) {
      int f4  = tid + i * 512;        // 0..2047 float4s
      int row = f4 >> 4;              // 0..127
      int c   = (f4 & 15) << 2;       // 0..60
      float4 v = qb[f4];
      ushort4 s;
      s.x = f2bf(v.x * 0.125f);
      s.y = f2bf(v.y * 0.125f);
      s.z = f2bf(v.z * 0.125f);
      s.w = f2bf(v.w * 0.125f);
      *(ushort4*)&Qs[row * LDSTRIDE + c] = s;
    }
  }
  // ---------------- Stage K windows [wi-1, wi, wi+1] as bf16 ----------------
  {
    const float* kb = k + (size_t)bh * 8192 * 64;
    #pragma unroll
    for (int i = 0; i < 12; ++i) {
      int f4  = tid + i * 512;        // 0..6143 float4s
      int row = f4 >> 4;              // 0..383 local key row
      int c   = (f4 & 15) << 2;
      int g   = wi * 128 - 128 + row; // global key row (may be OOB -> padded)
      ushort4 s{0, 0, 0, 0};
      if ((unsigned)g < 8192u) {
        float4 v = ((const float4*)(kb + (size_t)g * 64))[f4 & 15];
        s.x = f2bf(v.x); s.y = f2bf(v.y); s.z = f2bf(v.z); s.w = f2bf(v.w);
      }
      *(ushort4*)&Ks[row * LDSTRIDE + c] = s;
    }
  }
  __syncthreads();

  // ---------------- Per-wave WMMA GEMM: 16 rows x 192 cols ----------------
  const int wv    = tid >> 5;     // 0..15
  const int lane  = tid & 31;
  const int mtile = wv >> 1;      // 0..7  : 16-row strip
  const int nhalf = wv & 1;       // 0/1   : which 12 of 24 N-tiles
  const int l16   = lane & 15;
  const int half  = lane >> 4;

  // A fragments (16x32 bf16, ISA §7.12.2 layout), two K-chunks covering d=64
  U8 a0u, a1u;
  {
    const int arow = mtile * 16 + l16;
    #pragma unroll
    for (int v = 0; v < 8; ++v) {
      int kb = ((v & 3) << 1) + ((v >> 2) << 4) + (half << 3);
      a0u.u[v] = *(const unsigned*)&Qs[arow * LDSTRIDE + kb];
      a1u.u[v] = *(const unsigned*)&Qs[arow * LDSTRIDE + kb + 32];
    }
  }
  const v16bf A0 = __builtin_bit_cast(v16bf, a0u);
  const v16bf A1 = __builtin_bit_cast(v16bf, a1u);

  v8f acc[12];
  #pragma unroll
  for (int t = 0; t < 12; ++t) acc[t] = (v8f){0, 0, 0, 0, 0, 0, 0, 0};

  #pragma unroll
  for (int t = 0; t < 12; ++t) {
    const int brow = (nhalf * 12 + t) * 16 + l16;   // key row for this lane's N
    U8 b0u, b1u;
    #pragma unroll
    for (int v = 0; v < 8; ++v) {
      int kb = ((v & 3) << 1) + ((v >> 2) << 4) + (half << 3);  // B 32x16 bf16 layout (table k order, matches A)
      b0u.u[v] = *(const unsigned*)&Ks[brow * LDSTRIDE + kb];
      b1u.u[v] = *(const unsigned*)&Ks[brow * LDSTRIDE + kb + 32];
    }
    v16bf B0 = __builtin_bit_cast(v16bf, b0u);
    v16bf B1 = __builtin_bit_cast(v16bf, b1u);
    acc[t] = __builtin_amdgcn_wmma_f32_16x16x32_bf16(
        false, A0, false, B0, (short)0, acc[t], false, false);
    acc[t] = __builtin_amdgcn_wmma_f32_16x16x32_bf16(
        false, A1, false, B1, (short)0, acc[t], false, false);
  }

  // ---------------- Padding mask (only windows 0 and 63) ----------------
  if (wi == 0 || wi == 63) {
    #pragma unroll
    for (int t = 0; t < 12; ++t) {
      int col = (nhalf * 12 + t) * 16 + l16;        // 0..383
      bool masked = (wi == 0 && col < 128) || (wi == 63 && col >= 256);
      if (masked) {
        #pragma unroll
        for (int r = 0; r < 8; ++r) acc[t][r] = -3.402823466e38f;
      }
    }
  }

  // ---------------- Row softmax over 384 columns ----------------
  // row m = mtile*16 + half*8 + r ; its columns live in 16 lanes x 12 tiles.
  float pm[8];
  #pragma unroll
  for (int r = 0; r < 8; ++r) pm[r] = acc[0][r];
  #pragma unroll
  for (int t = 1; t < 12; ++t)
    #pragma unroll
    for (int r = 0; r < 8; ++r) pm[r] = fmaxf(pm[r], acc[t][r]);
  #pragma unroll
  for (int m = 1; m < 16; m <<= 1)
    #pragma unroll
    for (int r = 0; r < 8; ++r) pm[r] = fmaxf(pm[r], __shfl_xor(pm[r], m, 32));

  const int rowbase = mtile * 16 + half * 8;
  if (l16 == 0) {
    #pragma unroll
    for (int r = 0; r < 8; ++r) redmax[nhalf][rowbase + r] = pm[r];
  }
  __syncthreads();

  float gm[8];
  #pragma unroll
  for (int r = 0; r < 8; ++r)
    gm[r] = fmaxf(redmax[0][rowbase + r], redmax[1][rowbase + r]);

  float ps[8];
  #pragma unroll
  for (int r = 0; r < 8; ++r) ps[r] = 0.0f;
  #pragma unroll
  for (int t = 0; t < 12; ++t)
    #pragma unroll
    for (int r = 0; r < 8; ++r) {
      float e = __expf(acc[t][r] - gm[r]);
      acc[t][r] = e;
      ps[r] += e;
    }
  #pragma unroll
  for (int m = 1; m < 16; m <<= 1)
    #pragma unroll
    for (int r = 0; r < 8; ++r) ps[r] += __shfl_xor(ps[r], m, 32);

  if (l16 == 0) {
    #pragma unroll
    for (int r = 0; r < 8; ++r) redsum[nhalf][rowbase + r] = ps[r];
  }
  __syncthreads();

  float inv[8];
  #pragma unroll
  for (int r = 0; r < 8; ++r)
    inv[r] = 1.0f / (redsum[0][rowbase + r] + redsum[1][rowbase + r]);

  // ---------------- Store probabilities: out[bh][o][row][col] ----------------
  float* ob = out + (((size_t)bh * 62 + (size_t)o) * 128) * 384;
  #pragma unroll
  for (int t = 0; t < 12; ++t) {
    int col = (nhalf * 12 + t) * 16 + l16;
    #pragma unroll
    for (int r = 0; r < 8; ++r) {
      int row = rowbase + r;
      ob[(size_t)row * 384 + col] = acc[t][r] * inv[r];
    }
  }
}

extern "C" void kernel_launch(void* const* d_in, const int* in_sizes, int n_in,
                              void* d_out, int out_size, void* d_ws, size_t ws_size,
                              hipStream_t stream) {
  const float* q = (const float*)d_in[0];
  const float* k = (const float*)d_in[1];
  float* out = (float*)d_out;
  dim3 grid(62, 32);   // (remaining windows, B*H)
  local_attn_kernel<<<grid, dim3(512), 0, stream>>>(q, k, out);
}
